// MultiHeadLatentAttention_18227841204378
// MI455X (gfx1250) — matrix operand order repaired and checked by reading, not compiled, not run on hardware
//
#include <hip/hip_runtime.h>
#include <hip/hip_bf16.h>
#include <math.h>

typedef __bf16  v16bf __attribute__((ext_vector_type(16)));
typedef float   v8f   __attribute__((ext_vector_type(8)));

// ---------------------------------------------------------------------------
// WMMA helpers (gfx1250: D = A(16x32 bf16) * B(32x16 bf16) + C(16x16 f32))
// ---------------------------------------------------------------------------
__device__ __forceinline__ v8f wmma_bf16(v16bf a, v16bf b, v8f c) {
    return __builtin_amdgcn_wmma_f32_16x16x32_bf16(
        /*neg_a=*/false, a, /*neg_b=*/false, b,
        /*c_mod=*/(short)0, c, /*reuse_a=*/false, /*reuse_b=*/false);
}

// A-matrix 16x32 bf16 fragment: lane M = lane&15, hi = lane>>4.
// element e -> K = e + hi*16   (contiguous order, matches B fragments)
__device__ __forceinline__ v16bf load_afrag(const __bf16* row, int hi) {
    v16bf f;
#pragma unroll
    for (int e = 0; e < 16; e += 2) {
        const int k = e + hi * 16;
        f[e]     = row[k];
        f[e + 1] = row[k + 1];
    }
    return f;
}
// B-matrix 32x16 bf16 fragment: lane N = lane&15; element e -> K = e + hi*16
// => 16 contiguous bf16 (32B) per lane when B rows are K-contiguous in memory.

// Async global->LDS 16B copy (gfx1250 GLOBAL_LOAD_ASYNC_TO_LDS_B128, ASYNCcnt)
__device__ __forceinline__ void async_copy_b128(void* lds_ptr, const void* gptr) {
    const unsigned lds_off = (unsigned)(uintptr_t)lds_ptr;  // LDS aperture: addr[31:0]
    asm volatile("global_load_async_to_lds_b128 %0, %1, off"
                 :: "v"(lds_off), "v"(gptr)
                 : "memory");
}
__device__ __forceinline__ void async_wait_all() {
    asm volatile("s_wait_asynccnt 0" ::: "memory");
}

// ---------------------------------------------------------------------------
// Elementwise prep kernels
// ---------------------------------------------------------------------------
__global__ void k_f32_to_bf16(const float* __restrict__ in, __bf16* __restrict__ out, size_t n) {
    size_t i = (size_t)blockIdx.x * 256 + threadIdx.x;
    if (i < n) out[i] = (__bf16)in[i];
}

// W[K][N] fp32 -> Wt[N][K] bf16
__global__ void k_transpose_w(const float* __restrict__ W, __bf16* __restrict__ Wt, int K, int N) {
    size_t i = (size_t)blockIdx.x * 256 + threadIdx.x;
    if (i >= (size_t)K * N) return;
    int n = (int)(i / K), k = (int)(i % K);
    Wt[i] = (__bf16)W[(size_t)k * N + n];
}

// ---------------------------------------------------------------------------
// GEMM: C[M][N] = A[M][K](bf16) x Wt[N][K](bf16)^T, f32 accum.
// Block tile 128(M) x 64(N), Kc = 32, 8 waves (4Mx2N), wave tile 32x32.
// Staging uses async global->LDS copies.
// ---------------------------------------------------------------------------
__global__ __launch_bounds__(256) void k_gemm(
    const __bf16* __restrict__ A, const __bf16* __restrict__ Wt,
    void* __restrict__ outp, int M, int N, int K, int out_bf16, float scale)
{
    __shared__ __align__(32) __bf16 sA[128 * 32];
    __shared__ __align__(32) __bf16 sB[64 * 32];

    const int tid  = threadIdx.x;
    const int lane = tid & 31, wid = tid >> 5;
    const int hl = lane & 15, hi = lane >> 4;
    const int wm = (wid & 3) * 32;   // wave M offset in block tile
    const int wn = (wid >> 2) * 32;  // wave N offset in block tile
    const int m0 = blockIdx.y * 128;
    const int n0 = blockIdx.x * 64;

    // Per-thread staging coordinates (8 bf16 = 16B per thread per pass)
    const int ia0 = tid * 8;                 // A pass 0
    const int ia1 = tid * 8 + 2048;          // A pass 1
    const __bf16* gA0 = &A[(size_t)(m0 + (ia0 >> 5)) * K + (ia0 & 31)];
    const __bf16* gA1 = &A[(size_t)(m0 + (ia1 >> 5)) * K + (ia1 & 31)];
    const __bf16* gB0 = &Wt[(size_t)(n0 + (ia0 >> 5)) * K + (ia0 & 31)];

    v8f acc[2][2];
#pragma unroll
    for (int i = 0; i < 2; ++i)
#pragma unroll
        for (int j = 0; j < 2; ++j)
#pragma unroll
            for (int r = 0; r < 8; ++r) acc[i][j][r] = 0.0f;

    for (int k0 = 0; k0 < K; k0 += 32) {
        __syncthreads();
        // Async-stage A (128x32) and B (64x32) tiles: DMA path, no VGPR bounce
        async_copy_b128(&sA[ia0], gA0 + k0);
        async_copy_b128(&sA[ia1], gA1 + k0);
        async_copy_b128(&sB[ia0], gB0 + k0);
        // Prefetch next K-tile sources into cache (global_prefetch_b8)
        __builtin_prefetch(gA0 + k0 + 32, 0, 0);
        __builtin_prefetch(gA1 + k0 + 32, 0, 0);
        __builtin_prefetch(gB0 + k0 + 32, 0, 0);
        async_wait_all();
        __syncthreads();

        v16bf af[2], bfr[2];
#pragma unroll
        for (int i = 0; i < 2; ++i)
            af[i] = load_afrag(&sA[(wm + i * 16 + hl) * 32], hi);
#pragma unroll
        for (int j = 0; j < 2; ++j)
            bfr[j] = *(const v16bf*)&sB[(wn + j * 16 + hl) * 32 + hi * 16];
#pragma unroll
        for (int i = 0; i < 2; ++i)
#pragma unroll
            for (int j = 0; j < 2; ++j)
                acc[i][j] = wmma_bf16(af[i], bfr[j], acc[i][j]);
    }

    // Store: C element (M = r + hi*8 within tile, N = hl); branch hoisted.
    if (out_bf16) {
        __bf16* outb = (__bf16*)outp;
#pragma unroll
        for (int i = 0; i < 2; ++i)
#pragma unroll
            for (int j = 0; j < 2; ++j)
#pragma unroll
                for (int r = 0; r < 8; ++r) {
                    const int m = m0 + wm + i * 16 + r + hi * 8;
                    const int n = n0 + wn + j * 16 + hl;
                    outb[(size_t)m * N + n] = (__bf16)(acc[i][j][r] * scale);
                }
    } else {
        float* outf = (float*)outp;
#pragma unroll
        for (int i = 0; i < 2; ++i)
#pragma unroll
            for (int j = 0; j < 2; ++j)
#pragma unroll
                for (int r = 0; r < 8; ++r) {
                    const int m = m0 + wm + i * 16 + r + hi * 8;
                    const int n = n0 + wn + j * 16 + hl;
                    outf[(size_t)m * N + n] = acc[i][j][r] * scale;
                }
    }
}

// ---------------------------------------------------------------------------
// RMSNorm: one block per row, fp32 in -> bf16 out
// ---------------------------------------------------------------------------
__global__ __launch_bounds__(256) void k_rmsnorm(
    const float* __restrict__ in, const float* __restrict__ w,
    __bf16* __restrict__ out, int R)
{
    const int row = blockIdx.x;
    const float* x = in + (size_t)row * R;
    __shared__ float red[256];
    float ss = 0.0f;
    for (int i = threadIdx.x; i < R; i += 256) { const float v = x[i]; ss += v * v; }
    red[threadIdx.x] = ss;
    __syncthreads();
    for (int s = 128; s > 0; s >>= 1) {
        if (threadIdx.x < s) red[threadIdx.x] += red[threadIdx.x + s];
        __syncthreads();
    }
    const float inv = rsqrtf(red[0] / (float)R + 1e-5f);
    for (int i = threadIdx.x; i < R; i += 256)
        out[(size_t)row * R + i] = (__bf16)(x[i] * inv * w[i]);
}

// ---------------------------------------------------------------------------
// RoPE: q_rope fp32 [M][16*64] -> bf16 [M][16*64]   (per-head angles, 32 pairs)
// ---------------------------------------------------------------------------
__global__ void k_rope_q(const float* __restrict__ in, const float* __restrict__ cs,
                         const float* __restrict__ sn, __bf16* __restrict__ out, int M)
{
    size_t i = (size_t)blockIdx.x * 256 + threadIdx.x;      // over M*16*32
    if (i >= (size_t)M * 16 * 32) return;
    const int p = (int)(i % 32);
    const int h = (int)((i / 32) % 16);
    const size_t m = i / (32 * 16);
    const float c = cs[h * 32 + p], s = sn[h * 32 + p];
    const size_t base = m * 1024 + (size_t)h * 64 + 2 * p;
    const float re = in[base], im = in[base + 1];
    out[base]     = (__bf16)(re * c - im * s);
    out[base + 1] = (__bf16)(re * s + im * c);
}

// k_rope fp32 [M][64] -> per-head expanded bf16 [M][16*64]
__global__ void k_rope_k(const float* __restrict__ in, const float* __restrict__ cs,
                         const float* __restrict__ sn, __bf16* __restrict__ out, int M)
{
    size_t i = (size_t)blockIdx.x * 256 + threadIdx.x;      // over M*16*32
    if (i >= (size_t)M * 16 * 32) return;
    const int p = (int)(i % 32);
    const int h = (int)((i / 32) % 16);
    const size_t m = i / (32 * 16);
    const float c = cs[h * 32 + p], s = sn[h * 32 + p];
    const float re = in[m * 64 + 2 * p], im = in[m * 64 + 2 * p + 1];
    const size_t ob = m * 1024 + (size_t)h * 64 + 2 * p;
    out[ob]     = (__bf16)(re * c - im * s);
    out[ob + 1] = (__bf16)(re * s + im * c);
}

// V [b*S+s][h*128+d] -> Vt [((b*16+h)*128+d)*S + s]  (bf16)
__global__ void k_transpose_v(const __bf16* __restrict__ v, __bf16* __restrict__ vt, int S) {
    size_t i = (size_t)blockIdx.x * 256 + threadIdx.x;
    const size_t total = (size_t)2 * 16 * 128 * S;
    if (i >= total) return;
    const int s = (int)(i % S);
    size_t rest = i / S;
    const int d = (int)(rest % 128); rest /= 128;
    const int h = (int)(rest % 16);
    const int b = (int)(rest / 16);
    vt[i] = v[((size_t)(b * S + s)) * 2048 + h * 128 + d];
}

// ---------------------------------------------------------------------------
// Flash attention, causal. Block = 8 waves, each wave owns 16 query rows.
// Score tile via 6x WMMA over d=192; online softmax via half-wave shuffles;
// P -> LDS -> A-fragment; PV via 8x WMMA against Vt[b][h][d][s].
// ---------------------------------------------------------------------------
__global__ __launch_bounds__(256) void k_attn(
    const __bf16* __restrict__ qn, const __bf16* __restrict__ qr,
    const __bf16* __restrict__ kn, const __bf16* __restrict__ kr,
    const __bf16* __restrict__ vt, __bf16* __restrict__ out, int S)
{
    const int b = blockIdx.z, h = blockIdx.y;
    const int wid = threadIdx.x >> 5, lane = threadIdx.x & 31;
    const int hl = lane & 15, hi = lane >> 4;
    const int q0 = blockIdx.x * 128 + wid * 16;
    __shared__ __align__(32) __bf16 sP[8][16][32];

    // Q fragments (d = 192 in 6 chunks of 32)
    v16bf qf[6];
    {
        const __bf16* qrow_n = qn + ((size_t)(b * S + q0 + hl)) * 2048 + h * 128;
        const __bf16* qrow_r = qr + ((size_t)(b * S + q0 + hl)) * 1024 + h * 64;
#pragma unroll
        for (int ch = 0; ch < 4; ++ch) qf[ch] = load_afrag(qrow_n + ch * 32, hi);
#pragma unroll
        for (int ch = 4; ch < 6; ++ch) qf[ch] = load_afrag(qrow_r + (ch - 4) * 32, hi);
    }

    v8f o_acc[8];
    float mrun[8], lrun[8];
#pragma unroll
    for (int t = 0; t < 8; ++t)
#pragma unroll
        for (int r = 0; r < 8; ++r) o_acc[t][r] = 0.0f;
#pragma unroll
    for (int r = 0; r < 8; ++r) { mrun[r] = -1e30f; lrun[r] = 0.0f; }

    const float scale = 0.07216878364870323f;  // 1/sqrt(192)
    const int kv_end = q0 + 16;
    const __bf16* vbase = vt + ((size_t)(b * 16 + h) * 128) * S;

    for (int kvb = 0; kvb < kv_end; kvb += 32) {
        v8f s0, s1;
#pragma unroll
        for (int r = 0; r < 8; ++r) { s0[r] = 0.0f; s1[r] = 0.0f; }

        const __bf16* kn0 = kn + ((size_t)(b * S + kvb + hl)) * 2048 + h * 128;
        const __bf16* kr0 = kr + ((size_t)(b * S + kvb + hl)) * 1024 + h * 64;
        const __bf16* kn1 = kn0 + (size_t)16 * 2048;
        const __bf16* kr1 = kr0 + (size_t)16 * 1024;
#pragma unroll
        for (int ch = 0; ch < 6; ++ch) {
            const __bf16* p0 = (ch < 4) ? (kn0 + ch * 32 + hi * 16) : (kr0 + (ch - 4) * 32 + hi * 16);
            const __bf16* p1 = (ch < 4) ? (kn1 + ch * 32 + hi * 16) : (kr1 + (ch - 4) * 32 + hi * 16);
            s0 = wmma_bf16(qf[ch], *(const v16bf*)p0, s0);
            s1 = wmma_bf16(qf[ch], *(const v16bf*)p1, s1);
        }

#pragma unroll
        for (int r = 0; r < 8; ++r) {
            const int qa = q0 + r + hi * 8;
            float x0 = (kvb + hl      > qa) ? -1e30f : s0[r] * scale;
            float x1 = (kvb + 16 + hl > qa) ? -1e30f : s1[r] * scale;
            float mx = fmaxf(x0, x1);
#pragma unroll
            for (int m = 1; m < 16; m <<= 1) mx = fmaxf(mx, __shfl_xor(mx, m, 32));
            const float nm  = fmaxf(mrun[r], mx);
            const float p0v = __expf(x0 - nm);
            const float p1v = __expf(x1 - nm);
            float rs = p0v + p1v;
#pragma unroll
            for (int m = 1; m < 16; m <<= 1) rs += __shfl_xor(rs, m, 32);
            const float alpha = __expf(mrun[r] - nm);
            lrun[r] = lrun[r] * alpha + rs;
            mrun[r] = nm;
#pragma unroll
            for (int t = 0; t < 8; ++t) o_acc[t][r] *= alpha;
            sP[wid][r + hi * 8][hl]      = (__bf16)p0v;
            sP[wid][r + hi * 8][16 + hl] = (__bf16)p1v;
        }
        asm volatile("s_wait_dscnt 0" ::: "memory");
        const v16bf pf = load_afrag(&sP[wid][hl][0], hi);
#pragma unroll
        for (int t = 0; t < 8; ++t) {
            const __bf16* vp = vbase + (size_t)(t * 16 + hl) * S + kvb + hi * 16;
            o_acc[t] = wmma_bf16(pf, *(const v16bf*)vp, o_acc[t]);
        }
    }

#pragma unroll
    for (int r = 0; r < 8; ++r) {
        const float inv = 1.0f / lrun[r];
        const int sidx = q0 + r + hi * 8;
        __bf16* orow = out + (size_t)(b * S + sidx) * 2048 + h * 128;
#pragma unroll
        for (int t = 0; t < 8; ++t) orow[t * 16 + hl] = (__bf16)(o_acc[t][r] * inv);
    }
}

// ---------------------------------------------------------------------------
// Host launch
// ---------------------------------------------------------------------------
extern "C" void kernel_launch(void* const* d_in, const int* in_sizes, int n_in,
                              void* d_out, int out_size, void* d_ws, size_t ws_size,
                              hipStream_t stream) {
    (void)in_sizes; (void)n_in; (void)out_size; (void)ws_size;
    constexpr int B = 2, S = 2048, D = 2048, H = 16;
    constexpr int RQ = 1536, RKV = 512;
    constexpr int M = B * S;            // 4096
    constexpr int HDN = H * 128;        // 2048
    constexpr int HDR = H * 64;         // 1024

    const float* x      = (const float*)d_in[0];
    const float* fcos   = (const float*)d_in[2];
    const float* fsin   = (const float*)d_in[3];
    const float* w_cq   = (const float*)d_in[4];
    const float* qnw    = (const float*)d_in[5];
    const float* w_dqn  = (const float*)d_in[6];
    const float* w_dqr  = (const float*)d_in[7];
    const float* w_ckv  = (const float*)d_in[8];
    const float* kvnw   = (const float*)d_in[9];
    const float* w_dkn  = (const float*)d_in[10];
    const float* w_dv   = (const float*)d_in[11];
    const float* w_kr   = (const float*)d_in[12];
    const float* w_proj = (const float*)d_in[13];
    float* outp = (float*)d_out;

    char* base = (char*)d_ws;
    size_t off = 0;
    auto alloc = [&](size_t bytes) -> char* {
        char* p = base + off;
        off += (bytes + 255) & ~(size_t)255;
        return p;
    };
    __bf16* xb     = (__bf16*)alloc((size_t)M * D * 2);
    __bf16* wcqT   = (__bf16*)alloc((size_t)D * RQ * 2);
    __bf16* wdqnT  = (__bf16*)alloc((size_t)RQ * HDN * 2);
    __bf16* wdqrT  = (__bf16*)alloc((size_t)RQ * HDR * 2);
    __bf16* wckvT  = (__bf16*)alloc((size_t)D * RKV * 2);
    __bf16* wdknT  = (__bf16*)alloc((size_t)RKV * HDN * 2);
    __bf16* wdvT   = (__bf16*)alloc((size_t)RKV * HDN * 2);
    __bf16* wkrT   = (__bf16*)alloc((size_t)D * 64 * 2);
    __bf16* wprojT = (__bf16*)alloc((size_t)HDN * D * 2);
    float*  cq32   = (float*)alloc((size_t)M * RQ * 4);
    __bf16* nqb    = (__bf16*)alloc((size_t)M * RQ * 2);
    float*  ckv32  = (float*)alloc((size_t)M * RKV * 4);
    __bf16* nkvb   = (__bf16*)alloc((size_t)M * RKV * 2);
    __bf16* qnb    = (__bf16*)alloc((size_t)M * HDN * 2);
    float*  qr32   = (float*)alloc((size_t)M * HDR * 4);
    __bf16* qrb    = (__bf16*)alloc((size_t)M * HDR * 2);
    __bf16* knb    = (__bf16*)alloc((size_t)M * HDN * 2);
    __bf16* vb     = (__bf16*)alloc((size_t)M * HDN * 2);
    __bf16* vtb    = (__bf16*)alloc((size_t)M * HDN * 2);
    float*  kr32   = (float*)alloc((size_t)M * 64 * 4);
    __bf16* krb    = (__bf16*)alloc((size_t)M * HDR * 2);
    __bf16* attnb  = (__bf16*)alloc((size_t)M * HDN * 2);

    auto cdiv = [](size_t a, size_t b) { return (unsigned)((a + b - 1) / b); };

    // --- prep ---
    k_f32_to_bf16<<<cdiv((size_t)M * D, 256), 256, 0, stream>>>(x, xb, (size_t)M * D);
    k_transpose_w<<<cdiv((size_t)D * RQ, 256), 256, 0, stream>>>(w_cq, wcqT, D, RQ);
    k_transpose_w<<<cdiv((size_t)RQ * HDN, 256), 256, 0, stream>>>(w_dqn, wdqnT, RQ, HDN);
    k_transpose_w<<<cdiv((size_t)RQ * HDR, 256), 256, 0, stream>>>(w_dqr, wdqrT, RQ, HDR);
    k_transpose_w<<<cdiv((size_t)D * RKV, 256), 256, 0, stream>>>(w_ckv, wckvT, D, RKV);
    k_transpose_w<<<cdiv((size_t)RKV * HDN, 256), 256, 0, stream>>>(w_dkn, wdknT, RKV, HDN);
    k_transpose_w<<<cdiv((size_t)RKV * HDN, 256), 256, 0, stream>>>(w_dv, wdvT, RKV, HDN);
    k_transpose_w<<<cdiv((size_t)D * 64, 256), 256, 0, stream>>>(w_kr, wkrT, D, 64);
    k_transpose_w<<<cdiv((size_t)HDN * D, 256), 256, 0, stream>>>(w_proj, wprojT, HDN, D);

    // --- projections ---
    k_gemm<<<dim3(RQ / 64, M / 128), 256, 0, stream>>>(xb, wcqT, cq32, M, RQ, D, 0, 1.0f);
    k_rmsnorm<<<M, 256, 0, stream>>>(cq32, qnw, nqb, RQ);
    k_gemm<<<dim3(HDN / 64, M / 128), 256, 0, stream>>>(nqb, wdqnT, qnb, M, HDN, RQ, 1, 1.0f);
    k_gemm<<<dim3(HDR / 64, M / 128), 256, 0, stream>>>(nqb, wdqrT, qr32, M, HDR, RQ, 0, 1.0f);
    k_gemm<<<dim3(RKV / 64, M / 128), 256, 0, stream>>>(xb, wckvT, ckv32, M, RKV, D, 0, 1.0f);
    k_rmsnorm<<<M, 256, 0, stream>>>(ckv32, kvnw, nkvb, RKV);
    k_gemm<<<dim3(HDN / 64, M / 128), 256, 0, stream>>>(nkvb, wdknT, knb, M, HDN, RKV, 1, 1.0f);
    k_gemm<<<dim3(HDN / 64, M / 128), 256, 0, stream>>>(nkvb, wdvT, vb, M, HDN, RKV, 1,
                                                        0.02209708691207961f /* 1/sqrt(2048) */);
    k_gemm<<<dim3(64 / 64, M / 128), 256, 0, stream>>>(xb, wkrT, kr32, M, 64, D, 0, 1.0f);

    // --- rope + V transpose ---
    k_rope_q<<<cdiv((size_t)M * H * 32, 256), 256, 0, stream>>>(qr32, fcos, fsin, qrb, M);
    k_rope_k<<<cdiv((size_t)M * H * 32, 256), 256, 0, stream>>>(kr32, fcos, fsin, krb, M);
    k_transpose_v<<<cdiv((size_t)M * HDN, 256), 256, 0, stream>>>(vb, vtb, S);

    // --- attention ---
    k_attn<<<dim3(S / 128, H, B), 256, 0, stream>>>(qnb, qrb, knb, krb, vtb, attnb, S);

    // --- output projection (fp32 to d_out) ---
    k_gemm<<<dim3(D / 64, M / 128), 256, 0, stream>>>(attnb, wprojT, outp, M, D, HDN, 0, 1.0f);
}
